// EGGFloat_70136815943790
// MI455X (gfx1250) — 8-bit matrix operand order repaired and checked by reading, compiled, not run on hardware
//
#include <hip/hip_runtime.h>
#include <hip/hip_bf16.h>


typedef __attribute__((ext_vector_type(8))) int v8i;
typedef __attribute__((ext_vector_type(4))) int v4i;

#define NB   128
#define NS   1024
#define NH   256
#define NV   256
#define NL   6
#define NM   1024

#define NWG  16
#define TPB  256
#define WPB  (TPB / 32)
#define NTHREADS (NWG * TPB)

#if defined(__has_builtin)
#if __has_builtin(__builtin_amdgcn_global_load_async_to_lds_b128)
#define ASYNC_LDS 1
#endif
#if __has_builtin(__builtin_amdgcn_s_wait_asynccnt)
#define HAVE_WAIT_ASYNC 1
#endif
#endif

constexpr size_t OFF_SYNC = 0;
constexpr size_t OFF_EMB  = 256;
constexpr size_t OFF_LN1  = OFF_EMB  + 65536;
constexpr size_t OFF_LN2  = OFF_LN1  + 1536;
constexpr size_t OFF_WF   = OFF_LN2  + 1536;
constexpr size_t OFF_UF   = OFF_WF   + 393216;
constexpr size_t OFF_WH   = OFF_UF   + 393216;
constexpr size_t OFF_UH   = OFF_WH   + 393216;
constexpr size_t OFF_BF   = OFF_UH   + 393216;
constexpr size_t OFF_BH   = OFF_BF   + 1536;
constexpr size_t OFF_W1   = OFF_BH   + 1536;
constexpr size_t OFF_W2   = OFF_W1   + 1572864;
constexpr size_t OFF_LNO  = OFF_W2   + 1572864;
constexpr size_t OFF_HEAD = OFF_LNO  + 256;
constexpr size_t OFF_X    = OFF_HEAD + 65536;
constexpr size_t OFF_XM   = OFF_X    + 32768;
constexpr size_t OFF_FT   = OFF_XM   + 32768;
constexpr size_t OFF_GP   = OFF_FT   + 32768;
constexpr size_t OFF_HB   = OFF_GP   + 32768;
constexpr size_t OFF_ST   = OFF_HB   + 131072;
constexpr size_t WS_NEEDED = OFF_ST  + 196608;

__device__ __forceinline__ int clip8(int v) {
    return v < -127 ? -127 : (v > 127 ? 127 : v);
}

__device__ __forceinline__ int floordiv_pos(int a, int b) {
    int q = a / b, r = a % b;
    return (r != 0 && a < 0) ? q - 1 : q;
}

__device__ __forceinline__ v8i wmma_iu8(v8i a, v8i b, v8i c) {
    return __builtin_amdgcn_wmma_i32_16x16x64_iu8(true, a, true, b, c, false, false);
}

__device__ __forceinline__ void cp16(const signed char* g, signed char* l) {
#ifdef ASYNC_LDS
    __builtin_amdgcn_global_load_async_to_lds_b128(
        (__attribute__((address_space(1))) v4i*)g,
        (__attribute__((address_space(3))) v4i*)l, 0, 0);
#else
    *(int4*)l = *(const int4*)g;
#endif
}

__device__ __forceinline__ void wait_async_lds() {
#ifdef ASYNC_LDS
#ifdef HAVE_WAIT_ASYNC
    __builtin_amdgcn_s_wait_asynccnt(0);
#else
    asm volatile("s_wait_asynccnt 0x0" ::: "memory");
#endif
#endif
}

__device__ __forceinline__ void stage_B(const signed char* __restrict__ W,
                                        int ldr, int n0,
                                        signed char* __restrict__ sB) {
    const signed char* src = W + (size_t)n0 * ldr;
    const int total = 16 * ldr;
    for (int off = threadIdx.x * 16; off < total; off += TPB * 16)
        cp16(src + off, sB + off);
}

__device__ __forceinline__ v8i load_A(const signed char* __restrict__ A,
                                      int ldr, int m0, int k0, int lane) {
    const signed char* rp = A + (size_t)(m0 + (lane & 15)) * ldr + k0;
    const int khalf = (lane >> 4) << 3;
    v8i a;
#pragma unroll
    for (int j = 0; j < 8; ++j) {
        int k = ((j >> 1) << 4) + ((j & 1) << 2) + khalf;
        a[j] = *(const int*)(rp + k);
    }
    return a;
}

__device__ __forceinline__ v8i load_B_lds(const signed char* __restrict__ sB,
                                          int ldr, int k0, int lane) {
    const signed char* rp = sB + (size_t)(lane & 15) * ldr + k0;
    const int khalf = (lane >> 4) << 3;
    v8i b;
#pragma unroll
    for (int j = 0; j < 8; ++j) {
        int k = ((j >> 1) << 4) + ((j & 1) << 2) + khalf;
        b[j] = *(const int*)(rp + k);
    }
    return b;
}

__device__ __forceinline__ void grid_sync(volatile unsigned* cnt,
                                          volatile unsigned* gen) {
    __threadfence();
    __syncthreads();
    if (threadIdx.x == 0) {
        unsigned g = *gen;
        unsigned arrived = atomicAdd((unsigned*)cnt, 1u);
        if (arrived == (unsigned)(NWG - 1)) {
            *cnt = 0;
            __threadfence();
            atomicAdd((unsigned*)gen, 1u);
        } else {
            while (*gen == g) { __builtin_amdgcn_s_sleep(2); }
        }
    }
    __syncthreads();
}

__device__ __forceinline__ void ln_tile(const signed char* __restrict__ x,
                                        const signed char* __restrict__ w,
                                        signed char* __restrict__ sy,
                                        int m0, int lane) {
#pragma unroll 4
    for (int r = 0; r < 16; ++r) {
        const signed char* xr = x + (size_t)(m0 + r) * NH;
        int s = 0;
#pragma unroll
        for (int u = 0; u < 8; ++u) {
            int v = xr[lane * 8 + u];
            s += (v < 0) ? -v : v;
        }
#pragma unroll
        for (int off = 16; off > 0; off >>= 1) s += __shfl_xor(s, off, 32);
        int am = s >> 8;
        if (am < 1) am = 1;
#pragma unroll
        for (int u = 0; u < 8; ++u) {
            int c = lane * 8 + u;
            int rr = floordiv_pos((int)xr[c] * (int)w[c], am);
            sy[r * NH + c] = (signed char)clip8(rr);
        }
    }
}

__global__ void egg_quant_f32_i8(const float* __restrict__ in,
                                 signed char* __restrict__ out, int n) {
    int i = blockIdx.x * blockDim.x + threadIdx.x;
    if (i < n) {
        float r = rintf(in[i]);
        r = fminf(fmaxf(r, -127.f), 127.f);
        out[i] = (signed char)(int)r;
    }
}

__global__ void egg_init(signed char* __restrict__ ws) {
    int tid = blockIdx.x * blockDim.x + threadIdx.x;
    if (tid == 0) {
        ((unsigned*)(ws + OFF_SYNC))[0] = 0u;
        ((unsigned*)(ws + OFF_SYNC))[1] = 0u;
    }
    signed char* st = ws + OFF_ST;
    for (int i = tid; i < NL * NB * NH; i += gridDim.x * blockDim.x) st[i] = 0;
}

__global__ __launch_bounds__(TPB, 1)
void egg_main(const int* __restrict__ tokens,
              signed char* __restrict__ ws,
              float* __restrict__ out) {
    volatile unsigned* cnt = (volatile unsigned*)(ws + OFF_SYNC);
    volatile unsigned* gen = cnt + 1;

    const signed char* emb   = ws + OFF_EMB;
    const signed char* ln1q  = ws + OFF_LN1;
    const signed char* ln2q  = ws + OFF_LN2;
    const signed char* wfq   = ws + OFF_WF;
    const signed char* ufq   = ws + OFF_UF;
    const signed char* whq   = ws + OFF_WH;
    const signed char* uhq   = ws + OFF_UH;
    const signed char* bfq0  = ws + OFF_BF;
    const signed char* bhq0  = ws + OFF_BH;
    const signed char* w1q   = ws + OFF_W1;
    const signed char* w2q   = ws + OFF_W2;
    const signed char* lnoq  = ws + OFF_LNO;
    const signed char* headq = ws + OFF_HEAD;
    signed char* xbuf  = ws + OFF_X;
    signed char* xmid  = ws + OFF_XM;
    signed char* ftbuf = ws + OFF_FT;
    signed char* gpbuf = ws + OFF_GP;
    signed char* hbuf  = ws + OFF_HB;
    signed char* states = ws + OFF_ST;

    __shared__ __align__(16) signed char sY[WPB * 16 * NH];
    __shared__ __align__(16) signed char sB[16 * NM];

    const int lane = threadIdx.x & 31;
    const int wid  = threadIdx.x >> 5;
    const int tid  = blockIdx.x * TPB + threadIdx.x;
    const int m0   = wid * 16;
    const int n0g  = blockIdx.x * 16;
    signed char* myY = sY + wid * (16 * NH);

    for (int t = 0; t < NS; ++t) {
        for (int idx = tid; idx < NB * NH; idx += NTHREADS) {
            int b = idx >> 8, c = idx & 255;
            int tok = tokens[b * NS + t];
            xbuf[idx] = emb[(size_t)tok * NH + c];
        }
        for (int idx = tid; idx < NL * NB * NH; idx += NTHREADS) {
            int b = (idx >> 8) & 127;
            if (tokens[b * NS + t] == 0) states[idx] = 0;
        }
        if (t == 0) {
            for (size_t off = (size_t)tid * 64; off < 393216;
                 off += (size_t)NTHREADS * 64)
                __builtin_prefetch(wfq + off, 0, 0);
        }
        grid_sync(cnt, gen);

        for (int L = 0; L < NL; ++L) {
            const signed char* Wf  = wfq + (size_t)L * NH * NH;
            const signed char* Uf  = ufq + (size_t)L * NH * NH;
            const signed char* Wh  = whq + (size_t)L * NH * NH;
            const signed char* Uh  = uhq + (size_t)L * NH * NH;
            const signed char* bf  = bfq0 + (size_t)L * NH;
            const signed char* bh  = bhq0 + (size_t)L * NH;
            const signed char* w1L = w1q + (size_t)L * NM * NH;
            const signed char* w2L = w2q + (size_t)L * NH * NM;
            signed char* stL = states + (size_t)L * NB * NH;

            stage_B(Wf, NH, n0g, sB);
            stage_B(Uf, NH, n0g, sB + 16 * NH);
            ln_tile(xbuf, ln1q + (size_t)L * NH, myY, m0, lane);
            wait_async_lds();
            __syncthreads();
            {
                v8i a1 = {}, a2 = {};
                for (int k0 = 0; k0 < NH; k0 += 64) {
                    a1 = wmma_iu8(load_A(myY, NH, 0, k0, lane),
                                  load_B_lds(sB, NH, k0, lane), a1);
                    a2 = wmma_iu8(load_A(stL, NH, m0, k0, lane),
                                  load_B_lds(sB + 16 * NH, NH, k0, lane), a2);
                }
                int n  = n0g + (lane & 15);
                int mb = m0 + ((lane >> 4) << 3);
                int bb = bf[n];
#pragma unroll
                for (int j = 0; j < 8; ++j) {
                    int m  = mb + j;
                    int f  = clip8(clip8(a1[j] >> 8) + clip8(a2[j] >> 8) + bb);
                    int st = stL[(size_t)m * NH + n];
                    int gp = clip8(((f + 127) * st) >> 8);
                    ftbuf[(size_t)m * NH + n] = (signed char)f;
                    gpbuf[(size_t)m * NH + n] = (signed char)gp;
                }
            }
            grid_sync(cnt, gen);

            stage_B(Wh, NH, n0g, sB);
            stage_B(Uh, NH, n0g, sB + 16 * NH);
            ln_tile(xbuf, ln1q + (size_t)L * NH, myY, m0, lane);
            wait_async_lds();
            __syncthreads();
            {
                v8i a1 = {}, a2 = {};
                for (int k0 = 0; k0 < NH; k0 += 64) {
                    a1 = wmma_iu8(load_A(myY, NH, 0, k0, lane),
                                  load_B_lds(sB, NH, k0, lane), a1);
                    a2 = wmma_iu8(load_A(gpbuf, NH, m0, k0, lane),
                                  load_B_lds(sB + 16 * NH, NH, k0, lane), a2);
                }
                int n  = n0g + (lane & 15);
                int mb = m0 + ((lane >> 4) << 3);
                int bb = bh[n];
#pragma unroll
                for (int j = 0; j < 8; ++j) {
                    int m  = mb + j;
                    int ht = clip8(clip8(a1[j] >> 8) + clip8(a2[j] >> 8) + bb);
                    int st = stL[(size_t)m * NH + n];
                    int f  = ftbuf[(size_t)m * NH + n];
                    int ns = clip8(st + (((f + 127) * (ht - st)) >> 8));
                    stL[(size_t)m * NH + n] = (signed char)ns;
                    int res = xbuf[(size_t)m * NH + n];
                    xmid[(size_t)m * NH + n] = (signed char)clip8(ns + res);
                }
            }
            grid_sync(cnt, gen);

            ln_tile(xmid, ln2q + (size_t)L * NH, myY, m0, lane);
            __syncthreads();
            for (int c = blockIdx.x; c < NM / 16; c += NWG) {
                stage_B(w1L, NH, c * 16, sB);
                wait_async_lds();
                __syncthreads();
                v8i acc = {};
                for (int k0 = 0; k0 < NH; k0 += 64)
                    acc = wmma_iu8(load_A(myY, NH, 0, k0, lane),
                                   load_B_lds(sB, NH, k0, lane), acc);
                int n  = c * 16 + (lane & 15);
                int mb = m0 + ((lane >> 4) << 3);
#pragma unroll
                for (int j = 0; j < 8; ++j)
                    hbuf[(size_t)(mb + j) * NM + n] =
                        (signed char)clip8(acc[j] >> 8);
                __syncthreads();
            }
            grid_sync(cnt, gen);

            stage_B(w2L, NM, n0g, sB);
            wait_async_lds();
            __syncthreads();
            {
                v8i acc = {};
                for (int k0 = 0; k0 < NM; k0 += 64)
                    acc = wmma_iu8(load_A(hbuf, NM, m0, k0, lane),
                                   load_B_lds(sB, NM, k0, lane), acc);
                int n  = n0g + (lane & 15);
                int mb = m0 + ((lane >> 4) << 3);
#pragma unroll
                for (int j = 0; j < 8; ++j) {
                    int m = mb + j;
                    int v = clip8(acc[j] >> 9);
                    int r = xmid[(size_t)m * NH + n];
                    xbuf[(size_t)m * NH + n] = (signed char)clip8(v + r);
                }
            }
            grid_sync(cnt, gen);
        }

        stage_B(headq, NH, n0g, sB);
        ln_tile(xbuf, lnoq, myY, m0, lane);
        wait_async_lds();
        __syncthreads();
        {
            v8i acc = {};
            for (int k0 = 0; k0 < NH; k0 += 64)
                acc = wmma_iu8(load_A(myY, NH, 0, k0, lane),
                               load_B_lds(sB, NH, k0, lane), acc);
            int n  = n0g + (lane & 15);
            int mb = m0 + ((lane >> 4) << 3);
#pragma unroll
            for (int j = 0; j < 8; ++j) {
                int m = mb + j;
                out[((size_t)m * NS + t) * NV + n] = (float)clip8(acc[j] >> 8);
            }
        }
        grid_sync(cnt, gen);
    }

    const size_t base = (size_t)NB * NS * NV;
    for (int idx = tid; idx < NL * NB * NH; idx += NTHREADS)
        out[base + idx] = (float)states[idx];
}

extern "C" void kernel_launch(void* const* d_in, const int* in_sizes, int n_in,
                              void* d_out, int out_size, void* d_ws,
                              size_t ws_size, hipStream_t stream) {
    (void)in_sizes; (void)n_in; (void)out_size;
    if (ws_size < WS_NEEDED) return;

    signed char* ws = (signed char*)d_ws;
    auto q = [&](int i, size_t off, int n) {
        egg_quant_f32_i8<<<(n + 255) / 256, 256, 0, stream>>>(
            (const float*)d_in[i], ws + off, n);
    };
    q(1,  OFF_EMB,  NV * NH);
    q(2,  OFF_LN1,  NL * NH);
    q(3,  OFF_LN2,  NL * NH);
    q(4,  OFF_WF,   NL * NH * NH);
    q(5,  OFF_UF,   NL * NH * NH);
    q(6,  OFF_BF,   NL * NH);
    q(7,  OFF_WH,   NL * NH * NH);
    q(8,  OFF_UH,   NL * NH * NH);
    q(9,  OFF_BH,   NL * NH);
    q(10, OFF_W1,   NL * NM * NH);
    q(11, OFF_W2,   NL * NH * NM);
    q(12, OFF_LNO,  NH);
    q(13, OFF_HEAD, NV * NH);

    egg_init<<<64, 256, 0, stream>>>(ws);

    egg_main<<<NWG, TPB, 0, stream>>>(
        (const int*)d_in[0], ws, (float*)d_out);
}
